// BigramHashEmbedding_39943195853444
// MI455X (gfx1250) — matrix operand order repaired and checked by reading, not compiled, not run on hardware
//
#include <hip/hip_runtime.h>

// BigramHashEmbedding for MI455X (gfx1250, wave32).
// hash -> gather 128-f32 embed row -> GEMM (M=65536,N=512,K=128) -> scale.
// Store-bandwidth bound (~134MB out @ 23.3TB/s ~= 6-7us). Math uses
// split-bf16 (hi/lo) v_wmma_f32_16x16x32_bf16, 3 passes, f32 accumulate:
// ~f32 accuracy at ~3x fewer matrix issues than f32 WMMA, well under the
// memory roof.

#define HASH_SIZE 10240
#define PROJ_DIM  128
#define MODEL_DIM 512
#define SEQ_MASK  8191   // sequence length 8192 (power of 2)
#define LDSK      136    // 128 + 8 bf16 pad -> bank spread for B-frag reads

typedef __attribute__((ext_vector_type(16))) __bf16 v16bf;
typedef __attribute__((ext_vector_type(8)))  __bf16 v8bf;
typedef __attribute__((ext_vector_type(8)))  float  v8f;
typedef __attribute__((ext_vector_type(4)))  float  v4f;

__global__ __launch_bounds__(256)
void bigram_embed_proj_wmma(const int*   __restrict__ tok,
                            const float* __restrict__ embed,
                            const float* __restrict__ proj,
                            const float* __restrict__ scale_p,
                            float*       __restrict__ out)
{
    // proj chunk (64 cols x 128 K) as split bf16 in LDS
    __shared__ __bf16 Bhi[64 * LDSK];
    __shared__ __bf16 Blo[64 * LDSK];

    const int tid    = threadIdx.x;
    const int lane   = tid & 31;
    const int wave   = tid >> 5;          // 8 waves
    const int mBlock = blockIdx.x;        // 512 blocks x 128 tokens
    const int nChunk = blockIdx.y;        // 8 chunks x 64 columns

    const float scale = *scale_p;

    // ---- stage proj[nChunk*64 .. +64][0..128] into LDS as bf16 hi/lo ----
    // 64*128 f32 = 1024 groups of 8; 4 groups per thread, 16B LDS stores.
    for (int i = tid; i < 64 * (PROJ_DIM / 8); i += 256) {
        const int n  = i >> 4;            // 0..63
        const int k8 = (i & 15) * 8;      // 0,8,..,120
        const float* src = proj + (size_t)(nChunk * 64 + n) * PROJ_DIM + k8;
        v4f a = *(const v4f*)(src);
        v4f b = *(const v4f*)(src + 4);
        v8bf hi, lo;
#pragma unroll
        for (int j = 0; j < 4; ++j) {
            float x  = a[j];
            __bf16 xh = (__bf16)x;
            hi[j]     = xh;
            lo[j]     = (__bf16)(x - (float)xh);
            float y   = b[j];
            __bf16 yh = (__bf16)y;
            hi[4 + j] = yh;
            lo[4 + j] = (__bf16)(y - (float)yh);
        }
        *(v8bf*)&Bhi[n * LDSK + k8] = hi;
        *(v8bf*)&Blo[n * LDSK + k8] = lo;
    }
    __syncthreads();

    // ---- per-wave: 16 tokens. Branchless hash (keep EXEC all-1s for WMMA).
    const int tokBase = mBlock * 128 + wave * 16;
    const int t       = tokBase + (lane & 15);   // lanes L and L+16 share row
    const bool first  = ((t & SEQ_MASK) == 0);   // start of a sequence
    const unsigned t1 = (unsigned)tok[t];
    const unsigned t0 = (unsigned)tok[first ? t : (t - 1)]; // safe, discarded
    unsigned h = (36313u * t1 ^ 27191u * t0) % (unsigned)(HASH_SIZE - 1);
    h = first ? (unsigned)(HASH_SIZE - 1) : h;

    const float* erow = embed + (size_t)h * PROJ_DIM;
    const int half = lane >> 4;   // A frag: 0 -> K{c..c+7,c+16..c+23}, 1 -> +8

    // ---- gather A fragments: 4 k-steps of 16x32 bf16, hi/lo split ----
    v16bf Ahi[4], Alo[4];
#pragma unroll
    for (int kk = 0; kk < 4; ++kk) {
        const int c0 = kk * 32 + half * 8;
        v4f x0 = *(const v4f*)(erow + c0);
        v4f x1 = *(const v4f*)(erow + c0 + 4);
        v4f x2 = *(const v4f*)(erow + c0 + 16);
        v4f x3 = *(const v4f*)(erow + c0 + 20);
        float xs[16] = { x0[0],x0[1],x0[2],x0[3], x1[0],x1[1],x1[2],x1[3],
                         x2[0],x2[1],x2[2],x2[3], x3[0],x3[1],x3[2],x3[3] };
#pragma unroll
        for (int j = 0; j < 16; ++j) {
            __bf16 xh  = (__bf16)xs[j];
            Ahi[kk][j] = xh;
            Alo[kk][j] = (__bf16)(xs[j] - (float)xh);
        }
    }

    // ---- compute 4 N-tiles of 16 columns ----
    const int nCol = lane & 15;
#pragma unroll
    for (int nt = 0; nt < 4; ++nt) {
        v8f acc = {};
        const int nLoc = nt * 16 + nCol;            // column in LDS chunk
        const int kOff = half * 8;                  // B frag: half-wave K split
#pragma unroll
        for (int kk = 0; kk < 4; ++kk) {
            const __bf16* bp = &Bhi[nLoc * LDSK + kk * 32 + kOff];
            v8bf b0 = *(const v8bf*)(bp);
            v8bf b1 = *(const v8bf*)(bp + 16);
            v16bf bhi = __builtin_shufflevector(
                b0, b1, 0,1,2,3,4,5,6,7,8,9,10,11,12,13,14,15);
            const __bf16* lp = &Blo[nLoc * LDSK + kk * 32 + kOff];
            v8bf l0 = *(const v8bf*)(lp);
            v8bf l1 = *(const v8bf*)(lp + 16);
            v16bf blo = __builtin_shufflevector(
                l0, l1, 0,1,2,3,4,5,6,7,8,9,10,11,12,13,14,15);

            // (ahi+alo)(bhi+blo) ~= ahi*bhi + alo*bhi + ahi*blo
            acc = __builtin_amdgcn_wmma_f32_16x16x32_bf16(
                false, Ahi[kk], false, bhi, (short)0, acc, false, false);
            acc = __builtin_amdgcn_wmma_f32_16x16x32_bf16(
                false, Alo[kk], false, bhi, (short)0, acc, false, false);
            acc = __builtin_amdgcn_wmma_f32_16x16x32_bf16(
                false, Ahi[kk], false, blo, (short)0, acc, false, false);
        }
        // C/D layout: VGPR r -> row r (lanes 0-15), row r+8 (lanes 16-31)
        float* obase = out + (size_t)tokBase * MODEL_DIM
                           + nChunk * 64 + nt * 16 + nCol;
#pragma unroll
        for (int r = 0; r < 8; ++r) {
            const int row = r + half * 8;
            obase[(size_t)row * MODEL_DIM] = acc[r] * scale;
        }
    }
}

extern "C" void kernel_launch(void* const* d_in, const int* in_sizes, int n_in,
                              void* d_out, int out_size, void* d_ws, size_t ws_size,
                              hipStream_t stream) {
    const int*   tok    = (const int*)d_in[0];     // [8,8192] int32
    const float* embed  = (const float*)d_in[1];   // [10240,128] f32
    const float* proj   = (const float*)d_in[2];   // [512,128] f32
    const float* scale  = (const float*)d_in[3];   // scalar f32
    float*       out    = (float*)d_out;           // [8,8192,512] f32

    const int nTok    = in_sizes[0];               // 65536
    const int mBlocks = nTok / 128;                // 512
    const int nChunks = MODEL_DIM / 64;            // 8

    dim3 grid(mBlocks, nChunks, 1);
    bigram_embed_proj_wmma<<<grid, 256, 0, stream>>>(tok, embed, proj, scale, out);
}
